// NoisyGRUSeq2SeqWithFeatures_33784212750575
// MI455X (gfx1250) — matrix operand order repaired and checked by reading, not compiled, not run on hardware
//
#include <hip/hip_runtime.h>
#include <hip/hip_bf16.h>
#include <math.h>

// ---------------------------------------------------------------------------
// Types for CDNA5 WMMA (wave32): v_wmma_f32_16x16x32_bf16
// ---------------------------------------------------------------------------
typedef __bf16 bf16_t;
typedef __attribute__((ext_vector_type(16))) __bf16 v16bf;
typedef __attribute__((ext_vector_type(8)))  __bf16 v8bf;
typedef __attribute__((ext_vector_type(8)))  float  v8f;

#define BATCH   128
#define SEQ     64
#define HSTRIDE 3584   // h0|h1|h2 concatenated row stride (512+1024+2048)

// Block tile: M = 128 (8 waves x 16 rows), N strip = 128 (8 n-tiles).
// LDS double buffer for B fragments: 2 x 8 frags x 512 bf16 = 16 KB.
#define SMB_ELEMS (2 * 8 * 512)

// ---------------------------------------------------------------------------
// Fragment helpers
// ---------------------------------------------------------------------------
__device__ __forceinline__ v16bf cat8(v8bf lo, v8bf hi) {
  return __builtin_shufflevector(lo, hi, 0,1,2,3,4,5,6,7,8,9,10,11,12,13,14,15);
}

// A fragment (16x32 bf16) from row-major activations, row stride lda elems.
// lane l: row = mbase + (l&15); K offset +16 for lanes 16..31 (16 contiguous k).
__device__ __forceinline__ v16bf load_a(const bf16_t* A, int lda, int row, int k0) {
  const bf16_t* p = A + (size_t)row * lda + k0;
  return cat8(*(const v8bf*)p, *(const v8bf*)(p + 8));
}

__device__ __forceinline__ v8f wmma_bf16(v16bf a, v16bf b, v8f c) {
  return __builtin_amdgcn_wmma_f32_16x16x32_bf16(false, a, false, b, (short)0, c,
                                                 false, false);
}

__device__ __forceinline__ float sigmoidf_fast(float x) {
  return 1.0f / (1.0f + __expf(-x));
}

// Stage one packed B fragment (1 KB) to LDS: wave w handles n-tile w of strip.
__device__ __forceinline__ void stageB(const bf16_t* __restrict__ Bp, size_t frag,
                                       int w, int lane, bf16_t* smB, int p) {
  const bf16_t* src = Bp + ((frag << 5) + lane) * 16;   // lane owns 32 B
  v8bf x0 = *(const v8bf*)src;
  v8bf x1 = *(const v8bf*)(src + 8);
  bf16_t* d = smB + ((p * 8 + w) * 512) + lane * 16;
  *(v8bf*)d       = x0;
  *(v8bf*)(d + 8) = x1;
}

// ---------------------------------------------------------------------------
// Shared GEMM core: acc[nt] (nt = 0..7) accumulate A(128 x K) x B(K x 128strip)
// K is split into two segments: X (nkx chunks of 32) then H (nkh chunks).
// Wave w owns M-tile w; B fragments staged through LDS (ping-pong).
// ---------------------------------------------------------------------------
__device__ __forceinline__ void mm_core(v8f acc[8],
                                        const bf16_t* __restrict__ X, int ldx, int nkx,
                                        const bf16_t* __restrict__ H, int ldh, int nkh,
                                        const bf16_t* __restrict__ Bp, int nt0,
                                        int w, int lane, bf16_t* smB) {
  const int nkc  = nkx + nkh;
  const int arow = w * 16 + (lane & 15);
  const int koff = (lane >> 4) * 16;
  const size_t fbase = (size_t)(nt0 + w) * nkc;

  stageB(Bp, fbase, w, lane, smB, 0);
  __syncthreads();

  for (int kc = 0; kc < nkc; ++kc) {
    const int p = kc & 1;
    if (kc + 1 < nkc) stageB(Bp, fbase + kc + 1, w, lane, smB, p ^ 1);

    const bf16_t* A = (kc < nkx) ? X : H;
    const int lda   = (kc < nkx) ? ldx : ldh;
    const int k0    = ((kc < nkx) ? kc : (kc - nkx)) * 32 + koff;
    v16bf a = load_a(A, lda, arow, k0);

#pragma unroll
    for (int nt = 0; nt < 8; ++nt) {
      const bf16_t* bp = smB + ((p * 8 + nt) * 512) + lane * 16;
      v16bf b = cat8(*(const v8bf*)bp, *(const v8bf*)(bp + 8));
      acc[nt] = wmma_bf16(a, b, acc[nt]);
    }
    __syncthreads();
  }
}

// ---------------------------------------------------------------------------
// Weight pack: fp32 row-major (K x N) -> bf16 WMMA-B fragments.
// frag = nt*(K/32)+kc ; within frag lane l holds n = nt*16+(l&15),
// k = kc*32 + (l>>4)*16 + j  (j = 0..15 contiguous).
// ---------------------------------------------------------------------------
__global__ void pack_weight(const float* __restrict__ W, bf16_t* __restrict__ dst,
                            int K, int N) {
  size_t i = (size_t)blockIdx.x * 256 + threadIdx.x;
  size_t total = (size_t)K * N;
  if (i >= total) return;
  int j    = (int)(i & 15);
  int lane = (int)((i >> 4) & 31);
  size_t frag = i >> 9;
  int nkc = K >> 5;
  int nt = (int)(frag / nkc);
  int kc = (int)(frag % nkc);
  int n = nt * 16 + (lane & 15);
  int k = kc * 32 + (lane >> 4) * 16 + j;
  dst[i] = (bf16_t)W[(size_t)k * N + n];
}

// Embedding lookup -> bf16 [t][b][32]
__global__ void pack_emb(const int* __restrict__ seqs, const float* __restrict__ emb,
                         bf16_t* __restrict__ xemb) {
  int i = blockIdx.x * 256 + threadIdx.x;          // t*4096 + b*32 + k
  int k = i & 31;
  int b = (i >> 5) & 127;
  int t = i >> 12;
  int tok = seqs[b * SEQ + t];
  xemb[i] = (bf16_t)emb[tok * 32 + k];
}

__global__ void zero_bf16(bf16_t* p, int n) {
  int i = blockIdx.x * 256 + threadIdx.x;
  if (i < n) p[i] = (bf16_t)0.0f;
}

// ---------------------------------------------------------------------------
// Gates GEMM: pre_g = [x | h] @ Kg + bg  (N = 2*out)
// n <  out : rh[m][n]   = bf16( sigmoid(pre) * h[m][n] )
// n >= out : z [m][n-o] = sigmoid(pre)
// ---------------------------------------------------------------------------
__global__ __launch_bounds__(256)
void gru_gates(const bf16_t* __restrict__ X, int ldx, int nkx,
               const bf16_t* __restrict__ H, int nkh,      // stride HSTRIDE
               const bf16_t* __restrict__ KgP, const float* __restrict__ bg,
               bf16_t* __restrict__ rh, float* __restrict__ zbuf, int out) {
  __shared__ __align__(16) bf16_t smB[SMB_ELEMS];
  const int lane = threadIdx.x & 31;
  const int w    = threadIdx.x >> 5;
  const int nt0  = blockIdx.x * 8;

  v8f acc[8];
#pragma unroll
  for (int nt = 0; nt < 8; ++nt) {
    float bias = bg[(nt0 + nt) * 16 + (lane & 15)];
#pragma unroll
    for (int r = 0; r < 8; ++r) acc[nt][r] = bias;
  }

  mm_core(acc, X, ldx, nkx, H, HSTRIDE, nkh, KgP, nt0, w, lane, smB);

  const int koff  = (lane >> 4) * 8;
  const int rbase = w * 16 + koff;
#pragma unroll
  for (int nt = 0; nt < 8; ++nt) {
    int n = (nt0 + nt) * 16 + (lane & 15);
    if (n < out) {
#pragma unroll
      for (int r = 0; r < 8; ++r) {
        int m = rbase + r;
        float g  = sigmoidf_fast(acc[nt][r]);
        float hv = (float)H[(size_t)m * HSTRIDE + n];
        rh[(size_t)m * out + n] = (bf16_t)(g * hv);
      }
    } else {
      int nz = n - out;
#pragma unroll
      for (int r = 0; r < 8; ++r) {
        int m = rbase + r;
        zbuf[(size_t)m * out + nz] = sigmoidf_fast(acc[nt][r]);
      }
    }
  }
}

// ---------------------------------------------------------------------------
// Candidate GEMM: pre_c = [x | r*h] @ Kc + bc ; c = tanh(pre_c)
// h_new = z*h + (1-z)*c ; xnext = h_new (unmasked) ; H = mask ? h_new : h_old
// ---------------------------------------------------------------------------
__global__ __launch_bounds__(256)
void gru_cand(const bf16_t* __restrict__ X, int ldx, int nkx,
              const bf16_t* __restrict__ RH, int nkrh,
              const bf16_t* __restrict__ KcP, const float* __restrict__ bc,
              const float* __restrict__ zbuf,
              bf16_t* __restrict__ H,                     // stride HSTRIDE (r/w)
              bf16_t* __restrict__ xnext,
              const int* __restrict__ lens, int t, int out) {
  __shared__ __align__(16) bf16_t smB[SMB_ELEMS];
  const int lane = threadIdx.x & 31;
  const int w    = threadIdx.x >> 5;
  const int nt0  = blockIdx.x * 8;

  v8f acc[8];
#pragma unroll
  for (int nt = 0; nt < 8; ++nt) {
    float bias = bc[(nt0 + nt) * 16 + (lane & 15)];
#pragma unroll
    for (int r = 0; r < 8; ++r) acc[nt][r] = bias;
  }

  mm_core(acc, X, ldx, nkx, RH, out, nkrh, KcP, nt0, w, lane, smB);

  const int koff  = (lane >> 4) * 8;
  const int rbase = w * 16 + koff;
#pragma unroll
  for (int nt = 0; nt < 8; ++nt) {
    int n = (nt0 + nt) * 16 + (lane & 15);
#pragma unroll
    for (int r = 0; r < 8; ++r) {
      int m = rbase + r;
      float c  = tanhf(acc[nt][r]);
      float zv = zbuf[(size_t)m * out + n];
      float hv = (float)H[(size_t)m * HSTRIDE + n];
      float hn = zv * hv + (1.0f - zv) * c;
      xnext[(size_t)m * out + n] = (bf16_t)hn;
      H[(size_t)m * HSTRIDE + n] = (bf16_t)((t < lens[m]) ? hn : hv);
    }
  }
}

// ---------------------------------------------------------------------------
// Encoder: out = tanh( [h0|h1|h2] @ W_enc + b_enc )  (K=3584, N=512, f32 out)
// ---------------------------------------------------------------------------
__global__ __launch_bounds__(256)
void enc_proj(const bf16_t* __restrict__ H, const bf16_t* __restrict__ WP,
              const float* __restrict__ benc, float* __restrict__ out) {
  __shared__ __align__(16) bf16_t smB[SMB_ELEMS];
  const int lane = threadIdx.x & 31;
  const int w    = threadIdx.x >> 5;
  const int nt0  = blockIdx.x * 8;

  v8f acc[8];
#pragma unroll
  for (int nt = 0; nt < 8; ++nt) {
    float bias = benc[(nt0 + nt) * 16 + (lane & 15)];
#pragma unroll
    for (int r = 0; r < 8; ++r) acc[nt][r] = bias;
  }

  mm_core(acc, H, HSTRIDE, 0, H, HSTRIDE, HSTRIDE / 32, WP, nt0, w, lane, smB);

  const int koff  = (lane >> 4) * 8;
  const int rbase = w * 16 + koff;
#pragma unroll
  for (int nt = 0; nt < 8; ++nt) {
    int n = (nt0 + nt) * 16 + (lane & 15);
#pragma unroll
    for (int r = 0; r < 8; ++r)
      out[(size_t)(rbase + r) * 512 + n] = tanhf(acc[nt][r]);
  }
}

// ---------------------------------------------------------------------------
// Host launcher
// ---------------------------------------------------------------------------
extern "C" void kernel_launch(void* const* d_in, const int* in_sizes, int n_in,
                              void* d_out, int out_size, void* d_ws, size_t ws_size,
                              hipStream_t stream) {
  const int*   seqs = (const int*)d_in[0];
  const int*   lens = (const int*)d_in[1];
  const float* emb  = (const float*)d_in[2];
  const float* Kg[3] = {(const float*)d_in[3], (const float*)d_in[7],  (const float*)d_in[11]};
  const float* bg[3] = {(const float*)d_in[4], (const float*)d_in[8],  (const float*)d_in[12]};
  const float* Kc[3] = {(const float*)d_in[5], (const float*)d_in[9],  (const float*)d_in[13]};
  const float* bc[3] = {(const float*)d_in[6], (const float*)d_in[10], (const float*)d_in[14]};
  const float* Wenc = (const float*)d_in[15];
  const float* benc = (const float*)d_in[16];

  const int cin_[3]  = {32, 512, 1024};
  const int cout_[3] = {512, 1024, 2048};
  const int hoff[3]  = {0, 512, 1536};

  // ---- workspace layout (bf16 elements, 32B-aligned sub-buffers) ----
  bf16_t* w = (bf16_t*)d_ws;
  size_t off = 0;
  auto alloc = [&](size_t n) -> bf16_t* {
    bf16_t* p = w + off;
    off += (n + 15) & ~(size_t)15;
    return p;
  };
  bf16_t* pKg[3];
  bf16_t* pKc[3];
  for (int l = 0; l < 3; ++l) {
    pKg[l] = alloc((size_t)(cin_[l] + cout_[l]) * 2 * cout_[l]);
    pKc[l] = alloc((size_t)(cin_[l] + cout_[l]) * cout_[l]);
  }
  bf16_t* pWenc = alloc((size_t)HSTRIDE * 512);
  bf16_t* xemb  = alloc((size_t)SEQ * BATCH * 32);
  bf16_t* hbuf  = alloc((size_t)BATCH * HSTRIDE);
  bf16_t* act[3] = {alloc((size_t)BATCH * 512),
                    alloc((size_t)BATCH * 1024),
                    alloc((size_t)BATCH * 2048)};
  bf16_t* rh = alloc((size_t)BATCH * 2048);
  float*  zbuf = (float*)(w + off);                 // BATCH*2048 f32

  // ---- one-time (per launch) packing ----
  auto packW = [&](const float* W, bf16_t* dst, int K, int N) {
    size_t tot = (size_t)K * N;
    pack_weight<<<(unsigned)((tot + 255) / 256), 256, 0, stream>>>(W, dst, K, N);
  };
  for (int l = 0; l < 3; ++l) {
    packW(Kg[l], pKg[l], cin_[l] + cout_[l], 2 * cout_[l]);
    packW(Kc[l], pKc[l], cin_[l] + cout_[l], cout_[l]);
  }
  packW(Wenc, pWenc, HSTRIDE, 512);
  pack_emb<<<(SEQ * BATCH * 32) / 256, 256, 0, stream>>>(seqs, emb, xemb);
  zero_bf16<<<(BATCH * HSTRIDE + 255) / 256, 256, 0, stream>>>(hbuf, BATCH * HSTRIDE);

  // ---- recurrent scan ----
  for (int t = 0; t < SEQ; ++t) {
    const bf16_t* x = xemb + (size_t)t * BATCH * 32;
    int ldx = 32;
    for (int l = 0; l < 3; ++l) {
      int cin = cin_[l], cout = cout_[l];
      bf16_t* H = hbuf + hoff[l];
      int nkx = cin / 32, nkh = cout / 32;
      gru_gates<<<(2 * cout) / 128, 256, 0, stream>>>(x, ldx, nkx, H, nkh,
                                                      pKg[l], bg[l], rh, zbuf, cout);
      gru_cand<<<cout / 128, 256, 0, stream>>>(x, ldx, nkx, rh, nkh,
                                               pKc[l], bc[l], zbuf, H, act[l],
                                               lens, t, cout);
      x = act[l];
      ldx = cout;
    }
  }

  // ---- final projection ----
  enc_proj<<<512 / 128, 256, 0, stream>>>(hbuf, pWenc, benc, (float*)d_out);
}
